// VectorQuantizer_82514911691189
// MI455X (gfx1250) — matrix operand order repaired and checked by reading, not compiled, not run on hardware
//
#include <hip/hip_runtime.h>

typedef __attribute__((ext_vector_type(16))) __bf16 v16bf;
typedef __attribute__((ext_vector_type(8)))  float  v8f;

#define VQ_N      65536              // rows of x
#define VQ_D      64                 // embedding dim
#define VQ_K      1024               // codebook size
#define VQ_WTILES (VQ_N / 32)        // 2048 wave-tiles, 32 rows each
#define VQ_WPB    8                  // waves per block (256 threads)
#define VQ_BLOCKS (VQ_WTILES / VQ_WPB) // 256

#define WMMA_BF16(a, b, c) \
    __builtin_amdgcn_wmma_f32_16x16x32_bf16(false, (a), false, (b), (short)0, (c), false, false)

// ---------------------------------------------------------------------------
// Prep 1: split fp32 codebook into bf16 hi/lo parts (Markidis compensation)
// ---------------------------------------------------------------------------
__global__ void vq_prep_convert(const float* __restrict__ E,
                                __bf16* __restrict__ e_hi,
                                __bf16* __restrict__ e_lo, int n) {
    int i = blockIdx.x * blockDim.x + threadIdx.x;
    if (i < n) {
        float f = E[i];
        __bf16 h = (__bf16)f;
        e_hi[i] = h;
        e_lo[i] = (__bf16)(f - (float)h);
    }
}

// ---------------------------------------------------------------------------
// Prep 2: exact fp32 ||e_k||^2 per codebook row
// ---------------------------------------------------------------------------
__global__ void vq_prep_norm(const float* __restrict__ E,
                             float* __restrict__ e_norm2, int K) {
    int k = blockIdx.x * blockDim.x + threadIdx.x;
    if (k < K) {
        float s = 0.f;
        #pragma unroll 8
        for (int d = 0; d < VQ_D; ++d) {
            float v = E[k * VQ_D + d];
            s += v * v;
        }
        e_norm2[k] = s;
    }
}

// ---------------------------------------------------------------------------
// Main: each wave owns 32 rows (two 16-row A-tile sets) and scans the whole
// codebook; every B tile is shared by two accumulators (12 WMMA per 8 b128
// loads). Distance proxy per column:  ||e_n||^2 - 2 * (x . e_n).
// ---------------------------------------------------------------------------
__global__ __launch_bounds__(256) void vq_main(
    const float*  __restrict__ x,
    const float*  __restrict__ E,
    const __bf16* __restrict__ e_hi,
    const __bf16* __restrict__ e_lo,
    const float*  __restrict__ e_norm2,
    float*        __restrict__ out,
    float*        __restrict__ partials) {

    const int lane  = threadIdx.x & 31;
    const int wv    = threadIdx.x >> 5;
    const int wtile = blockIdx.x * VQ_WPB + wv;   // 0..2047, exact grid
    const int row_base = wtile << 5;              // 32 rows per wave
    const int nlane = lane & 15;
    const int hi16  = lane >> 4;                  // 0: lanes 0-15, 1: lanes 16-31
    const int koffA = hi16 << 3;                  // A k-offset: 0 or 8
    const int koffB = hi16 << 4;                  // B k-offset: 0 or 16

    // ---- Load two A tile sets (x rows) in contiguous k order (same as B), hi/lo bf16 ----
    // lanes 0-15: row=nlane, elems[0..15]=K(koffB..koffB+15) (contiguous, matches B)
    v16bf a_hi[2][2], a_lo[2][2];
    #pragma unroll
    for (int rs = 0; rs < 2; ++rs) {
        const float* xrow = x + (row_base + (rs << 4) + nlane) * VQ_D;
        #pragma unroll
        for (int kk = 0; kk < 2; ++kk) {
            v8f c0 = *(const v8f*)(xrow + kk * 32 + koffB);
            v8f c1 = *(const v8f*)(xrow + kk * 32 + koffB + 8);
            #pragma unroll
            for (int j = 0; j < 8; ++j) {
                float f0 = c0[j];
                __bf16 h0 = (__bf16)f0;
                a_hi[rs][kk][j] = h0;
                a_lo[rs][kk][j] = (__bf16)(f0 - (float)h0);
                float f1 = c1[j];
                __bf16 h1 = (__bf16)f1;
                a_hi[rs][kk][j + 8] = h1;
                a_lo[rs][kk][j + 8] = (__bf16)(f1 - (float)h1);
            }
        }
    }

    // ---- Scan 64 codebook tiles, lane-local argmin per row set ----
    float minval[2][8];
    int   minidx[2][8];
    #pragma unroll
    for (int rs = 0; rs < 2; ++rs)
        #pragma unroll
        for (int i = 0; i < 8; ++i) { minval[rs][i] = 3.4e38f; minidx[rs][i] = 0; }

    for (int t = 0; t < VQ_K / 16; ++t) {
        const int erow = (t << 4) + nlane;        // lane's codebook column
        v8f acc0 = {0.f, 0.f, 0.f, 0.f, 0.f, 0.f, 0.f, 0.f};
        v8f acc1 = {0.f, 0.f, 0.f, 0.f, 0.f, 0.f, 0.f, 0.f};
        #pragma unroll
        for (int kk = 0; kk < 2; ++kk) {
            // B layout: lane n holds column n; lanes 0-15 -> K 0..15,
            // lanes 16-31 -> K 16..31 (one contiguous 32B chunk per lane)
            const int bofs = erow * VQ_D + kk * 32 + koffB;
            v16bf bh = *(const v16bf*)(e_hi + bofs);
            v16bf bl = *(const v16bf*)(e_lo + bofs);
            acc0 = WMMA_BF16(a_hi[0][kk], bh, acc0);
            acc1 = WMMA_BF16(a_hi[1][kk], bh, acc1);
            acc0 = WMMA_BF16(a_lo[0][kk], bh, acc0);
            acc1 = WMMA_BF16(a_lo[1][kk], bh, acc1);
            acc0 = WMMA_BF16(a_hi[0][kk], bl, acc0);
            acc1 = WMMA_BF16(a_hi[1][kk], bl, acc1);
        }
        float en = e_norm2[erow];
        #pragma unroll
        for (int i = 0; i < 8; ++i) {
            float dp0 = en - 2.0f * acc0[i];
            bool b0 = dp0 < minval[0][i];         // strict: keeps lowest index
            minval[0][i] = b0 ? dp0 : minval[0][i];
            minidx[0][i] = b0 ? erow : minidx[0][i];
            float dp1 = en - 2.0f * acc1[i];
            bool b1 = dp1 < minval[1][i];
            minval[1][i] = b1 ? dp1 : minval[1][i];
            minidx[1][i] = b1 ? erow : minidx[1][i];
        }
    }

    // ---- Cross-lane argmin over the 16 n-lanes (both row halves at once) ----
    // element i: lanes 0-15 carry row m=i, lanes 16-31 carry row m=i+8
    #pragma unroll
    for (int m = 8; m >= 1; m >>= 1) {
        #pragma unroll
        for (int rs = 0; rs < 2; ++rs) {
            #pragma unroll
            for (int i = 0; i < 8; ++i) {
                float ov = __shfl_xor(minval[rs][i], m, 32);
                int   oi = __shfl_xor(minidx[rs][i], m, 32);
                bool take = (ov < minval[rs][i]) ||
                            (ov == minval[rs][i] && oi < minidx[rs][i]);
                minval[rs][i] = take ? ov : minval[rs][i];
                minidx[rs][i] = take ? oi : minidx[rs][i];
            }
        }
    }

    // ---- Broadcast the 32 winning indices to all lanes ----
    int rowIdx[32];
    #pragma unroll
    for (int r = 0; r < 8; ++r) {
        rowIdx[r]      = __shfl(minidx[0][r], 0, 32);   // rows  0..7
        rowIdx[r + 8]  = __shfl(minidx[0][r], 16, 32);  // rows  8..15
        rowIdx[r + 16] = __shfl(minidx[1][r], 0, 32);   // rows 16..23
        rowIdx[r + 24] = __shfl(minidx[1][r], 16, 32);  // rows 24..31
    }

    // ---- Gather fp32 embeddings, write output, accumulate loss partial ----
    float lsum = 0.f;
    #pragma unroll
    for (int r = 0; r < 32; ++r) {
        const int R = row_base + r;
        const int eoff = rowIdx[r] * VQ_D;
        const int xoff = R * VQ_D;
        float q0 = E[eoff + lane];
        float q1 = E[eoff + lane + 32];
        float x0 = x[xoff + lane];
        float x1 = x[xoff + lane + 32];
        out[xoff + lane]      = q0;   // straight-through forward == quantized
        out[xoff + lane + 32] = q1;
        float d0 = q0 - x0, d1 = q1 - x1;
        lsum += d0 * d0 + d1 * d1;
    }
    #pragma unroll
    for (int m = 16; m >= 1; m >>= 1) lsum += __shfl_xor(lsum, m, 32);
    if (lane == 0) partials[wtile] = lsum;
}

// ---------------------------------------------------------------------------
// Final: deterministic fixed-order reduction of loss partials
// loss = (1 + 0.25) * mean((q - x)^2)
// ---------------------------------------------------------------------------
__global__ void vq_loss_reduce(const float* __restrict__ partials, int n,
                               float* __restrict__ out_loss, float scale) {
    __shared__ float sm[256];
    float s = 0.f;
    for (int i = threadIdx.x; i < n; i += 256) s += partials[i];
    sm[threadIdx.x] = s;
    __syncthreads();
    for (int off = 128; off > 0; off >>= 1) {
        if ((int)threadIdx.x < off) sm[threadIdx.x] += sm[threadIdx.x + off];
        __syncthreads();
    }
    if (threadIdx.x == 0) *out_loss = sm[0] * scale;
}

// ---------------------------------------------------------------------------
extern "C" void kernel_launch(void* const* d_in, const int* in_sizes, int n_in,
                              void* d_out, int out_size, void* d_ws, size_t ws_size,
                              hipStream_t stream) {
    const float* x = (const float*)d_in[0];       // [65536, 64] fp32
    const float* E = (const float*)d_in[1];       // [1024, 64] fp32
    float* out = (float*)d_out;                   // 4194304 quantized + 1 loss

    // workspace layout
    char* ws = (char*)d_ws;
    __bf16* e_hi     = (__bf16*)(ws);             // 128 KB
    __bf16* e_lo     = (__bf16*)(ws + 131072);    // 128 KB
    float*  e_norm2  = (float*)(ws + 262144);     // 4 KB
    float*  partials = (float*)(ws + 266240);     // 8 KB

    vq_prep_convert<<<(VQ_K * VQ_D + 255) / 256, 256, 0, stream>>>(
        E, e_hi, e_lo, VQ_K * VQ_D);
    vq_prep_norm<<<(VQ_K + 255) / 256, 256, 0, stream>>>(E, e_norm2, VQ_K);
    vq_main<<<VQ_BLOCKS, 256, 0, stream>>>(
        x, E, e_hi, e_lo, e_norm2, out, partials);
    vq_loss_reduce<<<1, 256, 0, stream>>>(
        partials, VQ_WTILES, out + (out_size - 1),
        1.25f / (float)(VQ_N * VQ_D));
}
